// _METEnergy_18330920419669
// MI455X (gfx1250) — matrix operand order repaired and checked by reading, not compiled, not run on hardware
//
#include <hip/hip_runtime.h>
#include <hip/hip_bf16.h>
#include <math.h>

// Problem constants (match reference)
#define BQ 2048
#define LQ 256
#define DQ 64
#define HQ 4
#define DKQ 32
#define MQ 32
#define NQ 16
#define KQ 64

typedef __attribute__((ext_vector_type(16))) __bf16 v16bf;
typedef __attribute__((ext_vector_type(8)))  __bf16 v8bf;
typedef __attribute__((ext_vector_type(8)))  float  v8f;

static __device__ inline v8f wmma_bf16(v16bf a, v16bf b, v8f c) {
  // D = A(16x32 bf16) * B(32x16 bf16) + C(16x16 f32)
  return __builtin_amdgcn_wmma_f32_16x16x32_bf16(false, a, false, b, (short)0, c, false, false);
}

static __device__ inline v16bf cat8(v8bf lo, v8bf hi) {
  return __builtin_shufflevector(lo, hi, 0, 1, 2, 3, 4, 5, 6, 7, 8, 9, 10, 11, 12, 13, 14, 15);
}

// A-fragment: 16x32 tile, row-major storage (row = M), ISA layout:
// lane 0-15 rows M; elements 0..7 -> K=kb..kb+7, 8..15 -> K=16+kb..; kb=(lane>=16)*8
// Contiguous 8-element (16B) runs -> b128 loads.
static __device__ inline v16bf load_fragA_row(const __bf16* base, int row0, int k0, int ld, int lane) {
  const __bf16* p = base + (size_t)(row0 + (lane & 15)) * ld + k0 + ((lane >> 4) << 3);
  v8bf lo = *(const v8bf*)p;
  v8bf hi = *(const v8bf*)(p + 16);
  return cat8(lo, hi);
}

// B-fragment from transposed (N x K row-major) storage: lane 0-15 = col N,
// elements 0..7 -> K=kb..kb+7, 8..15 -> K=16+kb..; kb=(lane>=16)*8 -> two b128 loads.
static __device__ inline v16bf load_fragB_row(const __bf16* base, int n0, int k0, int ld, int lane) {
  const __bf16* p = base + (size_t)(n0 + (lane & 15)) * ld + k0 + ((lane >> 4) << 3);
  v8bf lo = *(const v8bf*)p;
  v8bf hi = *(const v8bf*)(p + 16);
  return cat8(lo, hi);
}

// C/D 16x16 f32: lane 0-15 -> N=lane, rows M=vgpr; lane 16-31 -> M=8+vgpr
static __device__ inline void store_tile_f(float* buf, int m0, int n0, int ld, v8f d, int lane) {
  int col   = n0 + (lane & 15);
  int rbase = m0 + ((lane >> 4) << 3);
#pragma unroll
  for (int r = 0; r < 8; ++r) buf[(rbase + r) * ld + col] = d[r];
}
static __device__ inline void store_tile_bf(__bf16* buf, int m0, int n0, int ld, v8f d, int lane) {
  int col   = n0 + (lane & 15);
  int rbase = m0 + ((lane >> 4) << 3);
#pragma unroll
  for (int r = 0; r < 8; ++r) buf[(rbase + r) * ld + col] = (__bf16)d[r];
}

// ---------------- prep kernel: batch-independent folds ----------------
__global__ __launch_bounds__(256)
void met_prep(const float* __restrict__ WQ, const float* __restrict__ WK,
              const float* __restrict__ Rv, const float* __restrict__ Fqv,
              const float* __restrict__ Ra, const float* __restrict__ Fqa,
              const float* __restrict__ Xiv, const float* __restrict__ Wcv,
              const float* __restrict__ Xia, const float* __restrict__ Wca,
              const float* __restrict__ wquad,
              __bf16* Pw, __bf16* Aw, __bf16* Ztv, __bf16* Zta,
              __bf16* Xnv, __bf16* Xna, float* logw) {
  int t = threadIdx.x;
  // P[mod][m][l] = Fq @ R   ([32,16]@[16,256])
  for (int i = t; i < 2 * MQ * LQ; i += 256) {
    int mod = i / (MQ * LQ);
    int rem = i % (MQ * LQ);
    int m = rem / LQ, l = rem % LQ;
    const float* Fq = mod ? Fqa : Fqv;
    const float* R  = mod ? Ra  : Rv;
    float s = 0.f;
    for (int n = 0; n < NQ; ++n) s += Fq[m * NQ + n] * R[n * LQ + l];
    Pw[i] = (__bf16)s;
  }
  // A_h[c][c'] = WQ_h^T @ WK_h  (4 x 64 x 64)
  for (int i = t; i < HQ * DQ * DQ; i += 256) {
    int h = i / (DQ * DQ);
    int rem = i % (DQ * DQ);
    int c = rem / DQ, cp = rem % DQ;
    float s = 0.f;
    for (int d = 0; d < DKQ; ++d) s += WQ[(h * DKQ + d) * DQ + c] * WK[(h * DKQ + d) * DQ + cp];
    Aw[i] = (__bf16)s;
  }
  // Normalized Xi rows and Zt = Xn @ Wc  (per modality)
  if (t < 2 * KQ) {
    int mod = t >> 6, k = t & 63;
    const float* Xi = mod ? Xia : Xiv;
    const float* Wc = mod ? Wca : Wcv;
    __bf16* Xn = mod ? Xna : Xnv;
    __bf16* Zt = mod ? Zta : Ztv;
    float nrm = 0.f;
    for (int d = 0; d < DQ; ++d) { float v = Xi[k * DQ + d]; nrm += v * v; }
    float inv = 1.f / fmaxf(sqrtf(nrm), 1e-12f);
    float xnr[DQ];
    for (int d = 0; d < DQ; ++d) { xnr[d] = Xi[k * DQ + d] * inv; Xn[k * DQ + d] = (__bf16)xnr[d]; }
    for (int c = 0; c < DQ; ++c) {
      float s = 0.f;
      for (int d = 0; d < DQ; ++d) s += xnr[d] * Wc[d * DQ + c];
      Zt[k * DQ + c] = (__bf16)s;
    }
  }
  if (t < MQ) logw[t] = logf(wquad[t]);
}

// ---------------- main fused kernel: one block per batch b ----------------
// LDS pool aliasing: gT (phase 1-2 only) overlays Ut/sA/sB (phase 3+ only).
#define POOL_BYTES (32768 + 18432 + 18432)   // Ut + sA + sB = 69632 >= gT 65536

__global__ __launch_bounds__(256)
void met_main(const float* __restrict__ x_v, const float* __restrict__ x_a,
              const float* __restrict__ lnvg, const float* __restrict__ lnvb,
              const float* __restrict__ lnag, const float* __restrict__ lnab,
              const float* __restrict__ lamv_p, const float* __restrict__ lama_p,
              const __bf16* __restrict__ Pw, const __bf16* __restrict__ Aw,
              const __bf16* __restrict__ Ztv, const __bf16* __restrict__ Zta,
              const __bf16* __restrict__ Xnv, const __bf16* __restrict__ Xna,
              const float* __restrict__ logw, float* __restrict__ blockE) {
  __shared__ __attribute__((aligned(16))) __bf16 gsh[2][LQ * DQ];   // 2x32KB layernormed acts
  __shared__ __attribute__((aligned(16))) unsigned char pool[POOL_BYTES];
  __shared__ __attribute__((aligned(16))) __bf16 Gq[2][MQ * DQ];    // 8KB  P @ g
  __shared__ float lw[MQ];

  __bf16* gT = (__bf16*)pool;                         // [2][64][256] bf16, phases 1-2
  __bf16 (*Ut)[MQ * DQ] = (__bf16(*)[MQ * DQ])pool;   // [8][32*64]  bf16, phases 3-4
  float* sA = (float*)(pool + 32768);                 // [256*18] f32, phases 4-5
  float* sB = (float*)(pool + 32768 + 18432);         // [256*18] f32, phases 4-5

  const int t = threadIdx.x, lane = t & 31, w = t >> 5;
  const int b = blockIdx.x;
  if (t < MQ) lw[t] = logw[t];

  // ---- Phase 1: LayerNorm rows (thread t owns row l=t); write g and gT ----
  {
    const float* xr_a = x_a + ((size_t)b * LQ + t) * DQ;
    __builtin_prefetch(xr_a, 0, 1);            // global_prefetch_b8: start x_a stream early
    __builtin_prefetch(xr_a + 32, 0, 1);
  }
#pragma unroll
  for (int mod = 0; mod < 2; ++mod) {
    const float* xp  = mod ? x_a : x_v;
    const float* gam = mod ? lnag : lnvg;
    const float* bet = mod ? lnab : lnvb;
    const float* xr  = xp + ((size_t)b * LQ + t) * DQ;
    float buf[DQ];
    float s1 = 0.f, s2 = 0.f;
#pragma unroll
    for (int i = 0; i < DQ / 4; ++i) {
      float4 q = ((const float4*)xr)[i];
      buf[4 * i + 0] = q.x; buf[4 * i + 1] = q.y; buf[4 * i + 2] = q.z; buf[4 * i + 3] = q.w;
      s1 += q.x + q.y + q.z + q.w;
      s2 += q.x * q.x + q.y * q.y + q.z * q.z + q.w * q.w;
    }
    float mu   = s1 * (1.f / DQ);
    float var  = s2 * (1.f / DQ) - mu * mu;
    float rstd = rsqrtf(var + 1e-5f);
    __bf16* gTm = gT + mod * (DQ * LQ);
#pragma unroll
    for (int i = 0; i < 8; ++i) {          // vectorized row store: 8x ds_store_b128
      v8bf v;
#pragma unroll
      for (int e = 0; e < 8; ++e) {
        int d = 8 * i + e;
        __bf16 gv = (__bf16)((buf[d] - mu) * rstd * gam[d] + bet[d]);
        v[e] = gv;
        gTm[d * LQ + t] = gv;              // transposed copy (scalar b16, stride 512B)
      }
      *(v8bf*)(&gsh[mod][t * DQ + 8 * i]) = v;
    }
  }
  __syncthreads();

  // ---- Phase 2: Gq[mod] = P_mod @ g_mod  ([32,256]@[256,64]); 8 output tiles -> 1/wave ----
  {
    int mt = w >> 2, nt = w & 3;
#pragma unroll
    for (int mod = 0; mod < 2; ++mod) {
      const __bf16* P   = Pw + mod * (MQ * LQ);
      const __bf16* gTm = gT + mod * (DQ * LQ);   // N x K storage (N=d, K=l)
      v8f acc = {};
#pragma unroll
      for (int kk = 0; kk < LQ / 32; ++kk) {
        v16bf a  = load_fragA_row(P, mt * 16, kk * 32, LQ, lane);
        v16bf bb = load_fragB_row(gTm, nt * 16, kk * 32, LQ, lane);
        acc = wmma_bf16(a, bb, acc);
      }
      store_tile_bf(Gq[mod], mt * 16, nt * 16, DQ, acc, lane);
    }
  }
  __syncthreads();   // gT dead from here; pool becomes Ut/sA/sB

  // ---- Phase 3: Ut[y*4+h] = Gq[y] @ A_h^T  ([32,64]@[64,64]); 64 tiles over 8 waves ----
#pragma unroll
  for (int i = 0; i < 8; ++i) {
    int j = w + 8 * i;
    int mat = j >> 3, tile = j & 7;
    int y = mat >> 2, h = mat & 3;
    int m0 = (tile >> 2) * 16, n0 = (tile & 3) * 16;
    v8f acc = {};
#pragma unroll
    for (int kk = 0; kk < 2; ++kk) {
      v16bf a  = load_fragA_row(Gq[y], m0, kk * 32, DQ, lane);
      v16bf bb = load_fragB_row(Aw + h * DQ * DQ, n0, kk * 32, DQ, lane);  // A_h is NxK storage
      acc = wmma_bf16(a, bb, acc);
    }
    store_tile_bf(Ut[mat], m0, n0, DQ, acc, lane);
  }
  __syncthreads();

  float en = 0.f;

  // ---- Phase 4: scores s = g_x @ Ut^T per (x, y, h); logsumexp over m=32 with logw ----
  for (int xm = 0; xm < 2; ++xm) {
    const __bf16* gx = gsh[xm];
    for (int c8 = 0; c8 < 8; ++c8) {
#pragma unroll
      for (int i = 0; i < 4; ++i) {
        int j = w + 8 * i;
        int mt = j & 15, half = j >> 4;   // half 0 -> cols m=0..15 (sA), 1 -> m=16..31 (sB)
        float* buf = half ? sB : sA;
        v8f acc = {};
#pragma unroll
        for (int kk = 0; kk < 2; ++kk) {
          v16bf a  = load_fragA_row(gx, mt * 16, kk * 32, DQ, lane);
          v16bf bb = load_fragB_row(Ut[c8], half * 16, kk * 32, DQ, lane);
          acc = wmma_bf16(a, bb, acc);
        }
        store_tile_f(buf, mt * 16, 0, 18, acc, lane);
      }
      __syncthreads();
      {
        float mx = -__builtin_inff();
#pragma unroll
        for (int jj = 0; jj < 16; ++jj) {
          mx = fmaxf(mx, sA[t * 18 + jj] + lw[jj]);
          mx = fmaxf(mx, sB[t * 18 + jj] + lw[16 + jj]);
        }
        float sum = 0.f;
#pragma unroll
        for (int jj = 0; jj < 16; ++jj) {
          sum += expf(sA[t * 18 + jj] + lw[jj] - mx);
          sum += expf(sB[t * 18 + jj] + lw[16 + jj] - mx);
        }
        en -= (mx + logf(sum));
      }
      __syncthreads();
    }
  }

  // ---- Phase 5: Hopfield. Hc = g_c @ Zt^T, Hm = g_m @ Xn^T; smooth rows of Hc; online LSE ----
  for (int p = 0; p < 2; ++p) {
    const __bf16* gm = gsh[p];         // p=0: mem = v, cross = a
    const __bf16* gc = gsh[1 - p];
    const __bf16* Zt = p ? Zta : Ztv;
    const __bf16* Xn = p ? Xna : Xnv;
    float lamr = p ? lama_p[0] : lamv_p[0];
    float lam  = fminf(fmaxf(lamr, 0.f), 1.f);
    float mrun = -__builtin_inff(), srun = 0.f;
    for (int kc = 0; kc < 4; ++kc) {   // k chunks of 16 (K=64 total)
#pragma unroll
      for (int i = 0; i < 4; ++i) {
        int j = w + 8 * i;
        int mt = j & 15, which = j >> 4;  // 0: Hc -> sA, 1: Hm -> sB
        const __bf16* Amat = which ? gm : gc;
        const __bf16* Bmat = which ? Xn : Zt;
        float* buf = which ? sB : sA;
        v8f acc = {};
#pragma unroll
        for (int kk = 0; kk < 2; ++kk) {
          v16bf a  = load_fragA_row(Amat, mt * 16, kk * 32, DQ, lane);
          v16bf bb = load_fragB_row(Bmat, kc * 16, kk * 32, DQ, lane);
          acc = wmma_bf16(a, bb, acc);
        }
        store_tile_f(buf, mt * 16, 0, 18, acc, lane);
      }
      __syncthreads();
      {
        int l = t;
        float inv_cnt = (l == 0 || l == LQ - 1) ? 0.5f : (1.f / 3.f);
#pragma unroll
        for (int jj = 0; jj < 16; ++jj) {
          float s = sA[l * 18 + jj];
          if (l > 0)      s += sA[(l - 1) * 18 + jj];
          if (l < LQ - 1) s += sA[(l + 1) * 18 + jj];
          float val = lam * (s * inv_cnt) + (1.f - lam) * sB[l * 18 + jj];
          if (val > mrun) { srun = srun * expf(mrun - val) + 1.f; mrun = val; }
          else            { srun += expf(val - mrun); }
        }
      }
      __syncthreads();
    }
    en -= (mrun + logf(srun));
  }

  // ---- block reduction -> per-block partial ----
  __syncthreads();
  sA[t] = en;
  __syncthreads();
  for (int s = 128; s > 0; s >>= 1) {
    if (t < s) sA[t] += sA[t + s];
    __syncthreads();
  }
  if (t == 0) blockE[b] = sA[0];
}

// ---------------- final deterministic sum ----------------
__global__ __launch_bounds__(256)
void met_final(const float* __restrict__ blockE, float* __restrict__ out) {
  __shared__ float r[256];
  int t = threadIdx.x;
  float s = 0.f;
  for (int i = t; i < BQ; i += 256) s += blockE[i];
  r[t] = s;
  __syncthreads();
  for (int k = 128; k > 0; k >>= 1) {
    if (t < k) r[t] += r[t + k];
    __syncthreads();
  }
  if (t == 0) out[0] = r[0];
}

extern "C" void kernel_launch(void* const* d_in, const int* in_sizes, int n_in,
                              void* d_out, int out_size, void* d_ws, size_t ws_size,
                              hipStream_t stream) {
  (void)in_sizes; (void)n_in; (void)out_size; (void)ws_size;
  const float* x_v  = (const float*)d_in[0];
  const float* x_a  = (const float*)d_in[1];
  const float* lnvg = (const float*)d_in[2];
  const float* lnvb = (const float*)d_in[3];
  const float* lnag = (const float*)d_in[4];
  const float* lnab = (const float*)d_in[5];
  const float* WQ   = (const float*)d_in[6];
  const float* WK   = (const float*)d_in[7];
  const float* lamv = (const float*)d_in[8];
  const float* Xiv  = (const float*)d_in[9];
  const float* Wcv  = (const float*)d_in[10];
  const float* lama = (const float*)d_in[11];
  const float* Xia  = (const float*)d_in[12];
  const float* Wca  = (const float*)d_in[13];
  const float* Rv   = (const float*)d_in[14];
  const float* Fqv  = (const float*)d_in[15];
  const float* Ra   = (const float*)d_in[16];
  const float* Fqa  = (const float*)d_in[17];
  const float* wq   = (const float*)d_in[18];

  char* ws = (char*)d_ws;                      // ws usage: ~107 KB
  __bf16* Pw   = (__bf16*)(ws + 0);            // [2][32][256] bf16 = 32768
  __bf16* Aw   = (__bf16*)(ws + 32768);        // [4][64][64]  bf16 = 32768
  __bf16* Ztv  = (__bf16*)(ws + 65536);        // [64][64] bf16 = 8192
  __bf16* Zta  = (__bf16*)(ws + 73728);
  __bf16* Xnv  = (__bf16*)(ws + 81920);
  __bf16* Xna  = (__bf16*)(ws + 90112);
  float*  logw = (float*)(ws + 98304);         // [32] f32
  float*  blkE = (float*)(ws + 98432);         // [2048] f32

  met_prep<<<1, 256, 0, stream>>>(WQ, WK, Rv, Fqv, Ra, Fqa, Xiv, Wcv, Xia, Wca, wq,
                                  Pw, Aw, Ztv, Zta, Xnv, Xna, logw);
  met_main<<<BQ, 256, 0, stream>>>(x_v, x_a, lnvg, lnvb, lnag, lnab, lamv, lama,
                                   Pw, Aw, Ztv, Zta, Xnv, Xna, logw, blkE);
  met_final<<<1, 256, 0, stream>>>(blkE, (float*)d_out);
}
